// QThreeLayerBN_88424786690236
// MI455X (gfx1250) — 8-bit matrix operand order repaired and checked by reading, compiled, not run on hardware
//
#include <hip/hip_runtime.h>
#include <math.h>

typedef int v8i __attribute__((ext_vector_type(8)));

constexpr int   BATCH   = 262144;
constexpr int   NBLK    = 2048;
constexpr float QMAXP   =  31.0f;
constexpr float QMINV   = -32.0f;
constexpr float BN_EPS  = 1e-5f;

struct Ctrl {
    unsigned amax_bits;
    float    s_act[4];
    float    sw[4][64];
    float    alpha[3][64];
    float    beta[3][64];
};
constexpr size_t CTRL_OFF = 0;
constexpr size_t WQ1_OFF  = 8192;
constexpr size_t WQ2_OFF  = WQ1_OFF + 64 * 64;
constexpr size_t WQ3_OFF  = WQ2_OFF + 32 * 64;
constexpr size_t WQ4_OFF  = WQ3_OFF + 32 * 64;
constexpr size_t PBSUM_OFF   = 64 * 1024;
constexpr size_t PBSUMSQ_OFF = PBSUM_OFF   + NBLK * 64 * 4;
constexpr size_t PBMAX_OFF   = PBSUMSQ_OFF + NBLK * 64 * 4;
constexpr size_t PBMIN_OFF   = PBMAX_OFF   + NBLK * 64 * 4;
constexpr size_t ZA_OFF   = 4u * 1024 * 1024;
constexpr size_t ZB_OFF   = ZA_OFF + (size_t)BATCH * 64 * 4;

__global__ void init_kernel(Ctrl* c) { c->amax_bits = 0u; }

__global__ void maxabs_kernel(const float* __restrict__ x, unsigned* __restrict__ amax, int n) {
    __shared__ unsigned red[256];
    unsigned m = 0u;
    for (int i = blockIdx.x * blockDim.x + threadIdx.x; i < n; i += gridDim.x * blockDim.x)
        m = max(m, __float_as_uint(fabsf(x[i])));
    red[threadIdx.x] = m;
    __syncthreads();
    for (int s = 128; s > 0; s >>= 1) {
        if ((int)threadIdx.x < s) red[threadIdx.x] = max(red[threadIdx.x], red[threadIdx.x + s]);
        __syncthreads();
    }
    if (threadIdx.x == 0) atomicMax(amax, red[0]);
}

__global__ void s0_kernel(Ctrl* c) {
    c->s_act[0] = fmaxf(__uint_as_float(c->amax_bits) * (1.0f / QMAXP), 1e-30f);
}

__global__ void wq_kernel(const float* __restrict__ w1, const float* __restrict__ w2,
                          const float* __restrict__ w3, const float* __restrict__ w4,
                          signed char* __restrict__ p1, signed char* __restrict__ p2,
                          signed char* __restrict__ p3, signed char* __restrict__ p4,
                          float* __restrict__ sw  ) {
    int t = threadIdx.x;
    const float* W; signed char* P; int K, N, c, li;
    if      (t < 64)  { li = 0; c = t;       W = w1; P = p1; K = 16; N = 64; }
    else if (t < 96)  { li = 1; c = t - 64;  W = w2; P = p2; K = 64; N = 32; }
    else if (t < 128) { li = 2; c = t - 96;  W = w3; P = p3; K = 32; N = 32; }
    else if (t < 144) { li = 3; c = t - 128; W = w4; P = p4; K = 32; N = 5;  }
    else return;
    if (c >= N) {
        for (int k = 0; k < 64; ++k) P[c * 64 + k] = 0;
        sw[li * 64 + c] = 0.0f;
        return;
    }
    float m = 0.0f;
    for (int k = 0; k < K; ++k) m = fmaxf(m, fabsf(W[c * K + k]));
    float s = m * (1.0f / QMAXP);
    sw[li * 64 + c] = s;
    float invs = (s > 0.0f) ? 1.0f / s : 0.0f;
    for (int k = 0; k < 64; ++k) {
        int q = 0;
        if (k < K) q = (int)fminf(fmaxf(rintf(W[c * K + k] * invs), QMINV), QMAXP);
        P[c * 64 + k] = (signed char)q;
    }
}

__global__ void bn_finalize_kernel(int N,
                                   const float* __restrict__ pb_sum, const float* __restrict__ pb_sumsq,
                                   const int* __restrict__ pb_max,  const int* __restrict__ pb_min,
                                   const float* __restrict__ sw_l, const float* __restrict__ g,
                                   const float* __restrict__ bt, const float* __restrict__ s_prev,
                                   float* __restrict__ alpha, float* __restrict__ beta,
                                   float* __restrict__ s_next) {
    __shared__ float red[64];
    int c = threadIdx.x;
    float hmax = 0.0f;
    if (c < N) {
        float sum = 0.0f, sumsq = 0.0f;
        int zmx = (int)0x80000000, zmn = 0x7fffffff;
        for (int i = 0; i < NBLK; ++i) {
            sum   += pb_sum[i * 64 + c];
            sumsq += pb_sumsq[i * 64 + c];
            zmx = max(zmx, pb_max[i * 64 + c]);
            zmn = min(zmn, pb_min[i * 64 + c]);
        }
        float sb   = sw_l[c] * (*s_prev);
        float mean = sum * (1.0f / (float)BATCH);
        float var  = fmaxf(sumsq * (1.0f / (float)BATCH) - mean * mean, 0.0f);
        float inv  = rsqrtf(sb * sb * var + BN_EPS);
        float a    = sb * inv * g[c];
        float be   = bt[c] - a * mean;
        alpha[c] = a; beta[c] = be;
        hmax = fmaxf(fmaxf(a * (float)zmx + be, 0.0f), fmaxf(a * (float)zmn + be, 0.0f));
    }
    red[threadIdx.x] = hmax;
    __syncthreads();
    for (int s = 32; s > 0; s >>= 1) {
        if ((int)threadIdx.x < s) red[threadIdx.x] = fmaxf(red[threadIdx.x], red[threadIdx.x + s]);
        __syncthreads();
    }
    if (threadIdx.x == 0) *s_next = fmaxf(red[0] * (1.0f / QMAXP), 1e-30f);
}

template<int K, bool FIRST>
__device__ inline v8i build_A(const float* __restrict__ xin, const int* __restrict__ zin,
                              const float* __restrict__ alpha, const float* __restrict__ beta,
                              float inv_s, int row, int hi) {
    v8i A = {0, 0, 0, 0, 0, 0, 0, 0};
#pragma unroll
    for (int v = 0; v < 8; ++v) {
        int kb = ((v >> 1) << 4) + ((v & 1) << 2) + (hi << 3);
        if (kb < K) {
            int q[4];
#pragma unroll
            for (int j = 0; j < 4; ++j) {
                int k = kb + j;
                float h;
                if (FIRST) {
                    h = xin[row * K + k];
                } else {
                    float z = (float)zin[row * K + k];
                    h = fmaxf(alpha[k] * z + beta[k], 0.0f);
                }
                q[j] = (int)fminf(fmaxf(rintf(h * inv_s), QMINV), QMAXP);
            }
            A[v] = (q[0] & 255) | ((q[1] & 255) << 8) | ((q[2] & 255) << 16) | ((q[3] & 255) << 24);
        }
    }
    return A;
}

__device__ inline v8i build_B(const signed char* __restrict__ wq, int ncol, int hi) {
    v8i Bm;
#pragma unroll
    for (int v = 0; v < 8; ++v) {
        int k = ((v >> 1) << 4) + ((v & 1) << 2) + (hi << 3);
        Bm[v] = *(const int*)(wq + ncol * 64 + k);
    }
    return Bm;
}

template<int K, int NOUT, bool FIRST>
__global__ __launch_bounds__(256)
void pass_kernel(const float* __restrict__ xin, const int* __restrict__ zin,
                 const signed char* __restrict__ wq,
                 const float* __restrict__ alpha, const float* __restrict__ beta,
                 const float* __restrict__ s_ptr,
                 int* __restrict__ zout,
                 float* __restrict__ pb_sum, float* __restrict__ pb_sumsq,
                 int* __restrict__ pb_max, int* __restrict__ pb_min) {
    __shared__ float wsum[8][NOUT], wsumsq[8][NOUT];
    __shared__ int   wmax[8][NOUT], wmin[8][NOUT];
    const int wave = threadIdx.x >> 5;
    const int lane = threadIdx.x & 31;
    const int hi   = lane >> 4;
    const int ncol = lane & 15;
    const int rowBase = (blockIdx.x * 8 + wave) * 16;
    const int row  = rowBase + (lane & 15);

    const float inv_s = 1.0f / (*s_ptr);
    v8i A = build_A<K, FIRST>(xin, zin, alpha, beta, inv_s, row, hi);

#pragma unroll
    for (int nt = 0; nt < NOUT / 16; ++nt) {
        v8i Bm = build_B(wq + nt * 16 * 64, ncol, hi);
        v8i C  = {0, 0, 0, 0, 0, 0, 0, 0};
        v8i D  = __builtin_amdgcn_wmma_i32_16x16x64_iu8(true, A, true, Bm, C, false, false);

        int n  = nt * 16 + ncol;
        int rr = rowBase + hi * 8;
        float psum = 0.0f, psumsq = 0.0f;
        int pmx = (int)0x80000000, pmn = 0x7fffffff;
#pragma unroll
        for (int r = 0; r < 8; ++r) {
            int z = D[r];
            zout[(rr + r) * NOUT + n] = z;
            float zf = (float)z;
            psum += zf; psumsq += zf * zf;
            pmx = max(pmx, z); pmn = min(pmn, z);
        }
        psum   += __shfl_xor(psum, 16, 32);
        psumsq += __shfl_xor(psumsq, 16, 32);
        pmx = max(pmx, __shfl_xor(pmx, 16, 32));
        pmn = min(pmn, __shfl_xor(pmn, 16, 32));
        if (hi == 0) {
            wsum[wave][n] = psum;  wsumsq[wave][n] = psumsq;
            wmax[wave][n] = pmx;   wmin[wave][n]   = pmn;
        }
    }
    __syncthreads();
    if ((int)threadIdx.x < NOUT) {
        int t = threadIdx.x;
        float bsum = 0.0f, bsumsq = 0.0f;
        int bmx = (int)0x80000000, bmn = 0x7fffffff;
#pragma unroll
        for (int w = 0; w < 8; ++w) {
            bsum   += wsum[w][t];
            bsumsq += wsumsq[w][t];
            bmx = max(bmx, wmax[w][t]);
            bmn = min(bmn, wmin[w][t]);
        }
        pb_sum  [blockIdx.x * 64 + t] = bsum;
        pb_sumsq[blockIdx.x * 64 + t] = bsumsq;
        pb_max  [blockIdx.x * 64 + t] = bmx;
        pb_min  [blockIdx.x * 64 + t] = bmn;
    }
}

__global__ __launch_bounds__(256)
void pass4_kernel(const int* __restrict__ zin, const signed char* __restrict__ wq,
                  const float* __restrict__ alpha, const float* __restrict__ beta,
                  const float* __restrict__ s_ptr, const float* __restrict__ sw4,
                  const float* __restrict__ b4, float* __restrict__ out) {
    const int lane = threadIdx.x & 31;
    const int wave = threadIdx.x >> 5;
    const int hi   = lane >> 4;
    const int ncol = lane & 15;
    const int rowBase = (blockIdx.x * 8 + wave) * 16;
    const int row  = rowBase + (lane & 15);

    const float s3 = *s_ptr;
    const float inv_s = 1.0f / s3;
    v8i A  = build_A<32, false>(nullptr, zin, alpha, beta, inv_s, row, hi);
    v8i Bm = build_B(wq, ncol, hi);
    v8i C  = {0, 0, 0, 0, 0, 0, 0, 0};
    v8i D  = __builtin_amdgcn_wmma_i32_16x16x64_iu8(true, A, true, Bm, C, false, false);

    const bool valid = ncol < 5;
    float sb = 0.0f, bi = 0.0f;
    if (valid) {
        sb = sw4[ncol] * s3;
        bi = fminf(fmaxf(rintf(b4[ncol] / sb), QMINV), QMAXP);
    }
    int rr = rowBase + hi * 8;
#pragma unroll
    for (int r = 0; r < 8; ++r) {
        float lz = valid ? ((float)D[r] + bi) * sb : -INFINITY;
        float m = lz;
#pragma unroll
        for (int msk = 1; msk < 16; msk <<= 1) m = fmaxf(m, __shfl_xor(m, msk, 32));
        float e = valid ? expf(lz - m) : 0.0f;
        float se = e;
#pragma unroll
        for (int msk = 1; msk < 16; msk <<= 1) se += __shfl_xor(se, msk, 32);
        if (valid) out[(rr + r) * 5 + ncol] = e / se;
    }
}

extern "C" void kernel_launch(void* const* d_in, const int* in_sizes, int n_in,
                              void* d_out, int out_size, void* d_ws, size_t ws_size,
                              hipStream_t stream) {
    const float* x   = (const float*)d_in[0];
    const float* w1  = (const float*)d_in[1];
    const float* w2  = (const float*)d_in[3];
    const float* w3  = (const float*)d_in[5];
    const float* w4  = (const float*)d_in[7];
    const float* b4  = (const float*)d_in[8];
    const float* g1  = (const float*)d_in[9];
    const float* bt1 = (const float*)d_in[10];
    const float* g2  = (const float*)d_in[11];
    const float* bt2 = (const float*)d_in[12];
    const float* g3  = (const float*)d_in[13];
    const float* bt3 = (const float*)d_in[14];

    char* ws = (char*)d_ws;
    Ctrl* ctrl = (Ctrl*)(ws + CTRL_OFF);
    signed char* p1 = (signed char*)(ws + WQ1_OFF);
    signed char* p2 = (signed char*)(ws + WQ2_OFF);
    signed char* p3 = (signed char*)(ws + WQ3_OFF);
    signed char* p4 = (signed char*)(ws + WQ4_OFF);
    float* pb_sum   = (float*)(ws + PBSUM_OFF);
    float* pb_sumsq = (float*)(ws + PBSUMSQ_OFF);
    int*   pb_max   = (int*)(ws + PBMAX_OFF);
    int*   pb_min   = (int*)(ws + PBMIN_OFF);
    int*   zA = (int*)(ws + ZA_OFF);
    int*   zB = (int*)(ws + ZB_OFF);
    float* outp = (float*)d_out;

    init_kernel<<<1, 1, 0, stream>>>(ctrl);
    wq_kernel<<<1, 256, 0, stream>>>(w1, w2, w3, w4, p1, p2, p3, p4, &ctrl->sw[0][0]);
    maxabs_kernel<<<1024, 256, 0, stream>>>(x, &ctrl->amax_bits, BATCH * 16);
    s0_kernel<<<1, 1, 0, stream>>>(ctrl);

    pass_kernel<16, 64, true><<<NBLK, 256, 0, stream>>>(
        x, nullptr, p1, nullptr, nullptr, &ctrl->s_act[0],
        zA, pb_sum, pb_sumsq, pb_max, pb_min);
    bn_finalize_kernel<<<1, 64, 0, stream>>>(
        64, pb_sum, pb_sumsq, pb_max, pb_min,
        &ctrl->sw[0][0], g1, bt1, &ctrl->s_act[0],
        ctrl->alpha[0], ctrl->beta[0], &ctrl->s_act[1]);

    pass_kernel<64, 32, false><<<NBLK, 256, 0, stream>>>(
        nullptr, zA, p2, ctrl->alpha[0], ctrl->beta[0], &ctrl->s_act[1],
        zB, pb_sum, pb_sumsq, pb_max, pb_min);
    bn_finalize_kernel<<<1, 64, 0, stream>>>(
        32, pb_sum, pb_sumsq, pb_max, pb_min,
        &ctrl->sw[1][0], g2, bt2, &ctrl->s_act[1],
        ctrl->alpha[1], ctrl->beta[1], &ctrl->s_act[2]);

    pass_kernel<32, 32, false><<<NBLK, 256, 0, stream>>>(
        nullptr, zB, p3, ctrl->alpha[1], ctrl->beta[1], &ctrl->s_act[2],
        zA, pb_sum, pb_sumsq, pb_max, pb_min);
    bn_finalize_kernel<<<1, 64, 0, stream>>>(
        32, pb_sum, pb_sumsq, pb_max, pb_min,
        &ctrl->sw[2][0], g3, bt3, &ctrl->s_act[2],
        ctrl->alpha[2], ctrl->beta[2], &ctrl->s_act[3]);

    pass4_kernel<<<NBLK, 256, 0, stream>>>(
        zA, p4, ctrl->alpha[2], ctrl->beta[2], &ctrl->s_act[3],
        &ctrl->sw[3][0], b4, outp);
}
